// CompressingTransformerModel_17970143166607
// MI455X (gfx1250) — matrix operand order repaired and checked by reading, not compiled, not run on hardware
//
#include <hip/hip_runtime.h>

#define USE_ASYNC_LDS 1

// ---------------------------------------------------------------------------
// Types for WMMA (CDNA5 / gfx1250, wave32)
// ---------------------------------------------------------------------------
typedef __bf16          bf16x16 __attribute__((ext_vector_type(16)));
typedef unsigned short  u16x16  __attribute__((ext_vector_type(16)));
typedef float           f32x8   __attribute__((ext_vector_type(8)));

union FragAB {
    u16x16  u;
    bf16x16 b;
};

static __device__ __forceinline__ unsigned short f32_to_bf16(float f) {
    unsigned int u = __float_as_uint(f);
    u += 0x7FFFu + ((u >> 16) & 1u);   // round-to-nearest-even
    return (unsigned short)(u >> 16);
}

// A-matrix (16x32 bf16) per-lane K index for fragment element j.
// Contiguous order (matches B fragments): K = hi*16 + j
static __device__ __forceinline__ int kmapA(int j, int hi) {
    return (hi ? 16 : 0) + j;
}

static __device__ __forceinline__ f32x8 wmma_bf16(const FragAB& a, const FragAB& b, f32x8 c) {
    return __builtin_amdgcn_wmma_f32_16x16x32_bf16(false, a.b, false, b.b,
                                                   (short)0, c, false, false);
}

static __device__ __forceinline__ f32x8 zero8() {
    f32x8 z = {0.f, 0.f, 0.f, 0.f, 0.f, 0.f, 0.f, 0.f};
    return z;
}

// ---------------------------------------------------------------------------
// VALU-only 16-lane reductions via DPP16 (no ds_bpermute round-trips).
// Sequence: quad xor1 (0xB1), quad xor2 (0x4E), row_half_mirror (lane^7, 0x141),
// row_mirror (lane^15, 0x140). After each step values are uniform over the
// combined group, so the mirror steps act as xor4/xor8.
// ---------------------------------------------------------------------------
template <int CTRL>
static __device__ __forceinline__ float dpp_mov(float x) {
    return __int_as_float(
        __builtin_amdgcn_update_dpp(0, __float_as_int(x), CTRL, 0xf, 0xf, true));
}

static __device__ __forceinline__ float row16_max(float v) {
    v = fmaxf(v, dpp_mov<0x0B1>(v));
    v = fmaxf(v, dpp_mov<0x04E>(v));
    v = fmaxf(v, dpp_mov<0x141>(v));
    v = fmaxf(v, dpp_mov<0x140>(v));
    return v;
}

static __device__ __forceinline__ float row16_sum(float v) {
    v += dpp_mov<0x0B1>(v);
    v += dpp_mov<0x04E>(v);
    v += dpp_mov<0x141>(v);
    v += dpp_mov<0x140>(v);
    return v;
}

// ---------------------------------------------------------------------------
// Elementwise helpers
// ---------------------------------------------------------------------------
__global__ void cvt_bf16(const float* __restrict__ in, unsigned short* __restrict__ out, size_t n) {
    size_t i = (size_t)blockIdx.x * blockDim.x + threadIdx.x;
    if (i < n) out[i] = f32_to_bf16(in[i]);
}

// Convert an f32 weight into WMMA-B fragment-ready bf16 layout:
//   out[(((kc*(N/16) + nt)*32 + lane)*16 + j)] = W[kc*32 + (lane>=16?16:0)+j][nt*16 + (lane&15)]
// TRANS=0: src is [K][N] row-major.  TRANS=1: src is [N][K] row-major (tied tok^T).
__global__ void cvt_fragB(const float* __restrict__ in, unsigned short* __restrict__ out,
                          int K, int N, int trans, size_t n) {
    size_t i = (size_t)blockIdx.x * blockDim.x + threadIdx.x;
    if (i >= n) return;
    int j = (int)(i & 15);
    int l = (int)((i >> 4) & 31);
    size_t rest = i >> 9;
    int NT = N >> 4;
    int nt = (int)(rest % (size_t)NT);
    size_t kc = rest / (size_t)NT;
    size_t k = kc * 32 + ((l >= 16) ? 16 : 0) + j;
    size_t nn = (size_t)nt * 16 + (l & 15);
    float v = trans ? in[nn * (size_t)K + k] : in[k * (size_t)N + nn];
    out[i] = f32_to_bf16(v);
}

__global__ void embed_kernel(const float* __restrict__ tok, const float* __restrict__ pos,
                             const int* __restrict__ idx, float* __restrict__ X, int T, int C) {
    int row = blockIdx.x;                 // 0 .. B*T-1
    int t = row % T;
    int id = idx[row];
    const float* tr = tok + (size_t)id * C;
    const float* pr = pos + (size_t)t * C;
    float* xr = X + (size_t)row * C;
    for (int c = threadIdx.x; c < C; c += blockDim.x) xr[c] = tr[c] + pr[c];
}

// LayerNorm over C=1024, one 256-thread block per row; bf16 output (GEMM staging)
__global__ void ln_bf16(const float* __restrict__ X, const float* __restrict__ g,
                        const float* __restrict__ bb, unsigned short* __restrict__ Y, int C) {
    __shared__ float red[256];
    const int row = blockIdx.x, tid = threadIdx.x;
    const float* x = X + (size_t)row * C;
    float v[4];
#pragma unroll
    for (int i = 0; i < 4; ++i) v[i] = x[tid + i * 256];
    float s = v[0] + v[1] + v[2] + v[3];
    red[tid] = s; __syncthreads();
    for (int off = 128; off; off >>= 1) { if (tid < off) red[tid] += red[tid + off]; __syncthreads(); }
    float mean = red[0] / (float)C;
    __syncthreads();
    float s2 = 0.f;
#pragma unroll
    for (int i = 0; i < 4; ++i) { float d = v[i] - mean; s2 += d * d; }
    red[tid] = s2; __syncthreads();
    for (int off = 128; off; off >>= 1) { if (tid < off) red[tid] += red[tid + off]; __syncthreads(); }
    float inv = rsqrtf(red[0] / (float)C + 1e-5f);
    unsigned short* y = Y + (size_t)row * C;
#pragma unroll
    for (int i = 0; i < 4; ++i) {
        int c = tid + i * 256;
        y[c] = f32_to_bf16((v[i] - mean) * inv * g[c] + bb[c]);
    }
}

// f32 [Btot*T, H*D] -> bf16 [Btot, H, T, D]   (row-major heads: Q, K)
__global__ void split_heads(const float* __restrict__ in, unsigned short* __restrict__ out,
                            int T, int H, int D, size_t n) {
    size_t i = (size_t)blockIdx.x * blockDim.x + threadIdx.x;
    if (i >= n) return;
    int HD = H * D;
    int c = (int)(i % (size_t)HD);
    size_t bt = i / (size_t)HD;
    int h = c / D, d = c % D;
    size_t b = bt / (size_t)T, t = bt % (size_t)T;
    out[(((b * H) + h) * (size_t)T + t) * D + d] = f32_to_bf16(in[i]);
}

// f32 [Btot*T, H*D] -> bf16 [Btot, H, D, T]   (transposed heads: V, for B-fragments)
__global__ void split_headsT(const float* __restrict__ in, unsigned short* __restrict__ out,
                             int T, int H, int D, size_t n) {
    size_t i = (size_t)blockIdx.x * blockDim.x + threadIdx.x;
    if (i >= n) return;
    int HD = H * D;
    int c = (int)(i % (size_t)HD);
    size_t bt = i / (size_t)HD;
    int h = c / D, d = c % D;
    size_t b = bt / (size_t)T, t = bt % (size_t)T;
    out[(((b * H) + h) * (size_t)D + d) * T + t] = f32_to_bf16(in[i]);
}

// ---------------------------------------------------------------------------
// WMMA GEMM: C[M,N] = A_bf16[M,K] @ Bfrag_bf16[K,N] (+bias) (+ReLU) (+residual)
// Block: 128 threads = 4 waves; block tile 64(M)x128(N); wave tile 32x64; K-step 32.
// A staged in LDS (async load-to-LDS); B read as fragment-ready 32B lane loads.
// M % 64 == 0, N % 128 == 0, K % 32 == 0 for all shapes used here.
// ---------------------------------------------------------------------------
template <int HAS_BIAS, int HAS_RES, int RELU>
__global__ void wmma_gemm(const unsigned short* __restrict__ A,
                          const unsigned short* __restrict__ Bf,
                          float* __restrict__ C,
                          int M, int N, int K,
                          const float* __restrict__ bias,
                          const float* __restrict__ residual) {
    __shared__ __align__(16) unsigned short As[64][40];  // 64x32, row stride 80B (16B-mult)

    const int tid  = threadIdx.x;
    const int lane = tid & 31, wid = tid >> 5;
    const int lo = lane & 15, hi = (lane >> 4) & 1;
    const int wm = (wid & 1) * 32;        // wave M offset within 64
    const int wn = (wid >> 1) * 64;       // wave N offset within 128
    const int m0 = blockIdx.y * 64, n0 = blockIdx.x * 128;
    const int NT = N >> 4;

    f32x8 acc[2][4];
#pragma unroll
    for (int mt = 0; mt < 2; ++mt)
#pragma unroll
        for (int nt = 0; nt < 4; ++nt) acc[mt][nt] = zero8();

    const int arow = tid >> 1, ahalf = (tid & 1) * 16;
    const size_t ntbase = (size_t)((n0 + wn) >> 4);

    for (int k0 = 0; k0 < K; k0 += 32) {
        const unsigned short* gsrc = A + (size_t)(m0 + arow) * K + k0 + ahalf;
#if USE_ASYNC_LDS
        {
            unsigned lds_off = (unsigned)(size_t)&As[arow][ahalf];
            asm volatile("global_load_async_to_lds_b128 %0, %2, off\n\t"
                         "global_load_async_to_lds_b128 %1, %3, off"
                         :
                         : "v"(lds_off), "v"(lds_off + 16u),
                           "v"(gsrc), "v"(gsrc + 8)
                         : "memory");
            asm volatile("s_wait_asynccnt 0" ::: "memory");
        }
#else
        {
            const uint4* src = (const uint4*)gsrc;
            uint4* dst = (uint4*)&As[arow][ahalf];
            dst[0] = src[0]; dst[1] = src[1];
        }
#endif
        if (k0 + 32 < K) __builtin_prefetch(gsrc + 32, 0, 0);   // global_prefetch_b8
        __syncthreads();

        FragAB a[2], b[4];
#pragma unroll
        for (int mt = 0; mt < 2; ++mt)
#pragma unroll
            for (int j = 0; j < 16; j += 2) {
                int k = kmapA(j, hi);
                a[mt].u[j]     = As[wm + mt * 16 + lo][k];
                a[mt].u[j + 1] = As[wm + mt * 16 + lo][k + 1];
            }
        const size_t kc = (size_t)(k0 >> 5);
#pragma unroll
        for (int nt = 0; nt < 4; ++nt) {
            size_t boff = ((kc * NT + ntbase + nt) * 32 + lane) * 16;
            b[nt].u = *(const u16x16*)(Bf + boff);
        }
#pragma unroll
        for (int mt = 0; mt < 2; ++mt)
#pragma unroll
            for (int nt = 0; nt < 4; ++nt)
                acc[mt][nt] = wmma_bf16(a[mt], b[nt], acc[mt][nt]);
        __syncthreads();
    }

#pragma unroll
    for (int nt = 0; nt < 4; ++nt) {
        size_t n = (size_t)(n0 + wn + nt * 16 + lo);
        float bv = HAS_BIAS ? bias[n] : 0.f;
#pragma unroll
        for (int mt = 0; mt < 2; ++mt)
#pragma unroll
            for (int r = 0; r < 8; ++r) {
                size_t m = (size_t)(m0 + wm + mt * 16 + r + hi * 8);
                float v = acc[mt][nt][r] + bv;
                if (RELU) v = fmaxf(v, 0.f);
                if (HAS_RES) v += residual[m * N + n];
                C[m * N + n] = v;
            }
    }
}

// ---------------------------------------------------------------------------
// Causal flash attention (inner block). One wave per 16-query tile.
// QH/KH: bf16 [B,H,T,64]; VH: bf16 [B,H,64,T]; O: f32 [B,T,H*64].
// ---------------------------------------------------------------------------
__global__ void flash_attn(const unsigned short* __restrict__ QH,
                           const unsigned short* __restrict__ KH,
                           const unsigned short* __restrict__ VH,
                           float* __restrict__ O, int T, int HC, float scale) {
    const int lane = threadIdx.x & 31;
    const int lo = lane & 15, hi = (lane >> 4) & 1;
    const int q0 = blockIdx.x * 16;
    const int h = blockIdx.y, b = blockIdx.z, H = gridDim.y;

    const unsigned short* Q = QH + ((size_t)(b * H + h) * T) * 64;
    const unsigned short* K = KH + ((size_t)(b * H + h) * T) * 64;
    const unsigned short* V = VH + ((size_t)(b * H + h) * 64) * T;

    FragAB aq[2];
#pragma unroll
    for (int c = 0; c < 2; ++c)
#pragma unroll
        for (int j = 0; j < 16; j += 2) {
            int k = kmapA(j, hi) + 32 * c;
            unsigned int w = *(const unsigned int*)&Q[(size_t)(q0 + lo) * 64 + k];
            aq[c].u[j]     = (unsigned short)(w & 0xffffu);
            aq[c].u[j + 1] = (unsigned short)(w >> 16);
        }

    f32x8 o[4];
    float rmax[8], rsum[8];
#pragma unroll
    for (int d = 0; d < 4; ++d) o[d] = zero8();
#pragma unroll
    for (int r = 0; r < 8; ++r) { rmax[r] = -1e30f; rsum[r] = 0.f; }

    __shared__ unsigned short Pl[16][32];

    for (int kt = 0; kt < q0 + 16; kt += 32) {
        f32x8 s0 = zero8(), s1 = zero8();
#pragma unroll
        for (int c = 0; c < 2; ++c) {
            FragAB bk0, bk1;
            bk0.u = *(const u16x16*)(K + (size_t)(kt + lo) * 64 + hi * 16 + 32 * c);
            bk1.u = *(const u16x16*)(K + (size_t)(kt + 16 + lo) * 64 + hi * 16 + 32 * c);
            s0 = wmma_bf16(aq[c], bk0, s0);
            s1 = wmma_bf16(aq[c], bk1, s1);
        }
        float corr[8];
#pragma unroll
        for (int r = 0; r < 8; ++r) {
            int mq = q0 + r + hi * 8;
            float a = s0[r] * scale, bbv = s1[r] * scale;
            if (kt + lo > mq)      a   = -1e30f;
            if (kt + 16 + lo > mq) bbv = -1e30f;
            float mloc = row16_max(fmaxf(a, bbv));
            float nm = fmaxf(rmax[r], mloc);
            float p0 = __expf(a - nm), p1 = __expf(bbv - nm);
            float ls = row16_sum(p0 + p1);
            float cr = __expf(rmax[r] - nm);
            rsum[r] = rsum[r] * cr + ls;
            rmax[r] = nm;
            corr[r] = cr;
            Pl[r + hi * 8][lo]      = f32_to_bf16(p0);
            Pl[r + hi * 8][16 + lo] = f32_to_bf16(p1);
        }
#pragma unroll
        for (int d = 0; d < 4; ++d)
#pragma unroll
            for (int r = 0; r < 8; ++r) o[d][r] *= corr[r];
        __syncthreads();
        FragAB pa;
#pragma unroll
        for (int j = 0; j < 16; j += 2) {
            int k = kmapA(j, hi);
            pa.u[j]     = Pl[lo][k];
            pa.u[j + 1] = Pl[lo][k + 1];
        }
#pragma unroll
        for (int dt = 0; dt < 4; ++dt) {
            FragAB bv;
            bv.u = *(const u16x16*)(V + (size_t)(dt * 16 + lo) * T + kt + hi * 16);
            o[dt] = wmma_bf16(pa, bv, o[dt]);
        }
        __syncthreads();
    }
#pragma unroll
    for (int dt = 0; dt < 4; ++dt)
#pragma unroll
        for (int r = 0; r < 8; ++r)
            O[((size_t)b * T + q0 + r + hi * 8) * HC + h * 64 + dt * 16 + lo] =
                o[dt][r] / rsum[r];
}

// ---------------------------------------------------------------------------
// Compress attention, pass 1: per-query softmax stats over keys r<=t.
// ---------------------------------------------------------------------------
__global__ void cattn_stats(const unsigned short* __restrict__ QH,
                            const unsigned short* __restrict__ KH,
                            float* __restrict__ SMAX, float* __restrict__ SINV,
                            int T, int TR, float scale) {
    const int lane = threadIdx.x & 31;
    const int lo = lane & 15, hi = (lane >> 4) & 1;
    const int q0 = blockIdx.x * 16;
    const int h = blockIdx.y, b = blockIdx.z, H = gridDim.y;

    const unsigned short* Q = QH + ((size_t)(b * H + h) * T) * 64;
    const unsigned short* K = KH + ((size_t)h * TR) * 64;

    FragAB aq[2];
#pragma unroll
    for (int c = 0; c < 2; ++c)
#pragma unroll
        for (int j = 0; j < 16; j += 2) {
            int k = kmapA(j, hi) + 32 * c;
            unsigned int w = *(const unsigned int*)&Q[(size_t)(q0 + lo) * 64 + k];
            aq[c].u[j]     = (unsigned short)(w & 0xffffu);
            aq[c].u[j + 1] = (unsigned short)(w >> 16);
        }

    float rmax[8], rsum[8];
#pragma unroll
    for (int r = 0; r < 8; ++r) { rmax[r] = -1e30f; rsum[r] = 0.f; }

    int kend = q0 + 16 < TR ? q0 + 16 : TR;
    for (int kt = 0; kt < kend; kt += 32) {
        f32x8 s0 = zero8(), s1 = zero8();
#pragma unroll
        for (int c = 0; c < 2; ++c) {
            FragAB bk0, bk1;
            bk0.u = *(const u16x16*)(K + (size_t)(kt + lo) * 64 + hi * 16 + 32 * c);
            bk1.u = *(const u16x16*)(K + (size_t)(kt + 16 + lo) * 64 + hi * 16 + 32 * c);
            s0 = wmma_bf16(aq[c], bk0, s0);
            s1 = wmma_bf16(aq[c], bk1, s1);
        }
#pragma unroll
        for (int r = 0; r < 8; ++r) {
            int mq = q0 + r + hi * 8;
            float a = s0[r] * scale, bbv = s1[r] * scale;
            if (kt + lo > mq)      a   = -1e30f;
            if (kt + 16 + lo > mq) bbv = -1e30f;
            float mloc = row16_max(fmaxf(a, bbv));
            float nm = fmaxf(rmax[r], mloc);
            float ls = row16_sum(__expf(a - nm) + __expf(bbv - nm));
            rsum[r] = rsum[r] * __expf(rmax[r] - nm) + ls;
            rmax[r] = nm;
        }
    }
    if (lo == 0) {
        size_t base = (size_t)(b * H + h) * T;
#pragma unroll
        for (int r = 0; r < 8; ++r) {
            size_t t = base + q0 + r + hi * 8;
            SMAX[t] = rmax[r];
            SINV[t] = 1.f / rsum[r];
        }
    }
}

// ---------------------------------------------------------------------------
// Compress attention, pass 2: out[r,:] = sum_t w[t,r] * V[t,:]
// VH: bf16 [B,H,64,T].  OUT: f32 [B, TR, H*64]
// ---------------------------------------------------------------------------
__global__ void cattn_out(const unsigned short* __restrict__ QH,
                          const unsigned short* __restrict__ KH,
                          const unsigned short* __restrict__ VH,
                          const float* __restrict__ SMAX, const float* __restrict__ SINV,
                          float* __restrict__ OUT, int T, int TR, int HC, float scale) {
    const int lane = threadIdx.x & 31;
    const int lo = lane & 15, hi = (lane >> 4) & 1;
    const int r0 = blockIdx.x * 16;
    const int h = blockIdx.y, b = blockIdx.z, H = gridDim.y;

    const unsigned short* Q = QH + ((size_t)(b * H + h) * T) * 64;
    const unsigned short* K = KH + ((size_t)h * TR) * 64;
    const unsigned short* V = VH + ((size_t)(b * H + h) * 64) * T;
    const float* smax = SMAX + (size_t)(b * H + h) * T;
    const float* sinv = SINV + (size_t)(b * H + h) * T;

    FragAB ak[2];
#pragma unroll
    for (int c = 0; c < 2; ++c)
#pragma unroll
        for (int j = 0; j < 16; j += 2) {
            int k = kmapA(j, hi) + 32 * c;
            unsigned int w = *(const unsigned int*)&K[(size_t)(r0 + lo) * 64 + k];
            ak[c].u[j]     = (unsigned short)(w & 0xffffu);
            ak[c].u[j + 1] = (unsigned short)(w >> 16);
        }

    f32x8 o[4];
#pragma unroll
    for (int d = 0; d < 4; ++d) o[d] = zero8();

    __shared__ unsigned short Pl[16][32];

    for (int tb = (r0 & ~31); tb < T; tb += 32) {
        f32x8 s0 = zero8(), s1 = zero8();
#pragma unroll
        for (int c = 0; c < 2; ++c) {
            FragAB bq0, bq1;
            bq0.u = *(const u16x16*)(Q + (size_t)(tb + lo) * 64 + hi * 16 + 32 * c);
            bq1.u = *(const u16x16*)(Q + (size_t)(tb + 16 + lo) * 64 + hi * 16 + 32 * c);
            s0 = wmma_bf16(ak[c], bq0, s0);
            s1 = wmma_bf16(ak[c], bq1, s1);
        }
        float mx0 = smax[tb + lo],      iv0 = sinv[tb + lo];
        float mx1 = smax[tb + 16 + lo], iv1 = sinv[tb + 16 + lo];
#pragma unroll
        for (int r = 0; r < 8; ++r) {
            int rr = r0 + r + hi * 8;   // reduced-key index (output row)
            float w0 = (rr <= tb + lo)      ? __expf(s0[r] * scale - mx0) * iv0 : 0.f;
            float w1 = (rr <= tb + 16 + lo) ? __expf(s1[r] * scale - mx1) * iv1 : 0.f;
            Pl[r + hi * 8][lo]      = f32_to_bf16(w0);
            Pl[r + hi * 8][16 + lo] = f32_to_bf16(w1);
        }
        __syncthreads();
        FragAB pa;
#pragma unroll
        for (int j = 0; j < 16; j += 2) {
            int k = kmapA(j, hi);
            pa.u[j]     = Pl[lo][k];
            pa.u[j + 1] = Pl[lo][k + 1];
        }
#pragma unroll
        for (int dt = 0; dt < 4; ++dt) {
            FragAB bv;
            bv.u = *(const u16x16*)(V + (size_t)(dt * 16 + lo) * T + tb + hi * 16);
            o[dt] = wmma_bf16(pa, bv, o[dt]);
        }
        __syncthreads();
    }
#pragma unroll
    for (int dt = 0; dt < 4; ++dt)
#pragma unroll
        for (int r = 0; r < 8; ++r)
            OUT[((size_t)b * TR + r0 + r + hi * 8) * HC + h * 64 + dt * 16 + lo] = o[dt][r];
}

// ---------------------------------------------------------------------------
// Host orchestration
// ---------------------------------------------------------------------------
extern "C" void kernel_launch(void* const* d_in, const int* in_sizes, int n_in,
                              void* d_out, int out_size, void* d_ws, size_t ws_size,
                              hipStream_t stream) {
    (void)in_sizes; (void)n_in; (void)out_size; (void)ws_size;
    const int B = 2, H = 16, C = 1024, HID = 4096, VOC = 32000;
    const float scale = 0.03125f;   // 1024^-0.5

    // Input index map (setup_inputs insertion order, nested dicts flattened):
    // 0 tok, 1 pos, blocks1 @2 (2x27), 56 ln_mid_g, 57 ln_mid_b, 58 mid_w,
    // 59 mid_b, blocks2 @60 (2x27), 114 ln_out_g, 115 ln_out_b, 116 out_b, 117 idx
    auto Pf = [&](int i) { return (const float*)d_in[i]; };
    const int* idx = (const int*)d_in[117];

    // Workspace arena (~225 MB required)
    char* ws = (char*)d_ws;
    float*          X    = (float*)(ws);
    float*          X2   = (float*)(ws + ((size_t)16 << 20));
    unsigned short* LNB  = (unsigned short*)(ws + ((size_t)32 << 20));
    float*          SCR  = (float*)(ws + ((size_t)40 << 20));
    unsigned short* SCRB = (unsigned short*)(ws + ((size_t)104 << 20));
    unsigned short* QHB  = (unsigned short*)(ws + ((size_t)136 << 20));
    unsigned short* KHB  = (unsigned short*)(ws + ((size_t)144 << 20));
    unsigned short* VHB  = (unsigned short*)(ws + ((size_t)152 << 20));
    unsigned short* WBF  = (unsigned short*)(ws + ((size_t)160 << 20));
    float*          SMAX = (float*)(ws + ((size_t)224 << 20));
    float*          SINV = (float*)(ws + ((size_t)224 << 20) + ((size_t)1 << 19));

    auto CVT = [&](unsigned short* dst, const float* src, size_t n) {
        cvt_bf16<<<dim3((unsigned)((n + 255) / 256)), dim3(256), 0, stream>>>(src, dst, n);
    };
    auto CVTW = [&](unsigned short* dst, const float* src, int K, int N, int trans) {
        size_t n = (size_t)K * N;
        cvt_fragB<<<dim3((unsigned)((n + 255) / 256)), dim3(256), 0, stream>>>(src, dst, K, N, trans, n);
    };
    auto GEMM = [&](float* Cp, const unsigned short* Ap, const unsigned short* Bp,
                    int M, int N, int K, const float* bias, const float* res, int relu) {
        dim3 g(N / 128, M / 64), blk(128);
        if (bias && res)
            wmma_gemm<1, 1, 0><<<g, blk, 0, stream>>>(Ap, Bp, Cp, M, N, K, bias, res);
        else if (bias && relu)
            wmma_gemm<1, 0, 1><<<g, blk, 0, stream>>>(Ap, Bp, Cp, M, N, K, bias, nullptr);
        else if (bias)
            wmma_gemm<1, 0, 0><<<g, blk, 0, stream>>>(Ap, Bp, Cp, M, N, K, bias, nullptr);
        else
            wmma_gemm<0, 0, 0><<<g, blk, 0, stream>>>(Ap, Bp, Cp, M, N, K, nullptr, nullptr);
    };
    auto LNK = [&](unsigned short* dst, const float* x, const float* g, const float* b, int rows) {
        ln_bf16<<<dim3(rows), dim3(256), 0, stream>>>(x, g, b, dst, C);
    };
    auto SPLIT = [&](unsigned short* dst, const float* src, int Bc, int Tc) {
        size_t n = (size_t)Bc * Tc * C;
        split_heads<<<dim3((unsigned)((n + 255) / 256)), dim3(256), 0, stream>>>(src, dst, Tc, H, 64, n);
    };
    auto SPLITT = [&](unsigned short* dst, const float* src, int Bc, int Tc) {
        size_t n = (size_t)Bc * Tc * C;
        split_headsT<<<dim3((unsigned)((n + 255) / 256)), dim3(256), 0, stream>>>(src, dst, Tc, H, 64, n);
    };

    int T = 2048;
    embed_kernel<<<dim3(B * T), dim3(256), 0, stream>>>(Pf(0), Pf(1), idx, X, T, C);

    float* x = X;
    float* xalt = X2;
    const int bases[4] = {2, 29, 60, 87};
    const int trs[4]   = {1024, 512, 1024, 2048};
    // cblock-relative offsets:
    // inner: 0 ln1g,1 ln1b,2 wq,3 wk,4 wv,5 wp,6 bp,7 ln2g,8 ln2b,9 w1,10 b1,11 w2,12 b2
    // 13 rpe,14 ln1g,15 ln1b,16 wq,17 wk,18 wv,19 wp,20 bp,21 ln2g,22 ln2b,23 w1,24 b1,25 w2,26 b2

    for (int bi = 0; bi < 4; ++bi) {
        if (bi == 2) {   // mid projection between blocks1 and blocks2
            int BT = B * T;
            LNK(LNB, x, Pf(56), Pf(57), BT);
            CVTW(WBF, Pf(58), C, C, 0);
            GEMM(xalt, LNB, WBF, BT, C, C, Pf(59), nullptr, 0);
            float* tmp = x; x = xalt; xalt = tmp;
        }
        const int base = bases[bi], TR = trs[bi];
        int BT = B * T;

        // ---- inner block: x += self_attn(ln1(x)) ----
        LNK(LNB, x, Pf(base + 0), Pf(base + 1), BT);
        CVTW(WBF, Pf(base + 2), C, C, 0);
        GEMM(SCR, LNB, WBF, BT, C, C, nullptr, nullptr, 0);
        SPLIT(QHB, SCR, B, T);
        CVTW(WBF, Pf(base + 3), C, C, 0);
        GEMM(SCR, LNB, WBF, BT, C, C, nullptr, nullptr, 0);
        SPLIT(KHB, SCR, B, T);
        CVTW(WBF, Pf(base + 4), C, C, 0);
        GEMM(SCR, LNB, WBF, BT, C, C, nullptr, nullptr, 0);
        SPLITT(VHB, SCR, B, T);
        flash_attn<<<dim3(T / 16, H, B), dim3(32), 0, stream>>>(QHB, KHB, VHB, SCR, T, C, scale);
        CVT(SCRB, SCR, (size_t)BT * C);
        CVTW(WBF, Pf(base + 5), C, C, 0);
        GEMM(x, SCRB, WBF, BT, C, C, Pf(base + 6), x, 0);

        // ---- inner block: x += ff(ln2(x)) ----
        LNK(LNB, x, Pf(base + 7), Pf(base + 8), BT);
        CVTW(WBF, Pf(base + 9), C, HID, 0);
        GEMM(SCR, LNB, WBF, BT, HID, C, Pf(base + 10), nullptr, 1);
        CVT(SCRB, SCR, (size_t)BT * HID);
        CVTW(WBF, Pf(base + 11), HID, C, 0);
        GEMM(x, SCRB, WBF, BT, C, HID, Pf(base + 12), x, 0);

        // ---- compress attention: x = cattn(ln1(x), rpe)  [T -> TR] ----
        LNK(LNB, x, Pf(base + 14), Pf(base + 15), BT);
        CVTW(WBF, Pf(base + 16), C, C, 0);
        GEMM(SCR, LNB, WBF, BT, C, C, nullptr, nullptr, 0);
        SPLIT(QHB, SCR, B, T);
        CVT(SCRB, Pf(base + 13), (size_t)TR * C);     // rpe -> bf16
        CVTW(WBF, Pf(base + 17), C, C, 0);
        GEMM(SCR, SCRB, WBF, TR, C, C, nullptr, nullptr, 0);
        SPLIT(KHB, SCR, 1, TR);
        CVTW(WBF, Pf(base + 18), C, C, 0);
        GEMM(SCR, LNB, WBF, BT, C, C, nullptr, nullptr, 0);
        SPLITT(VHB, SCR, B, T);
        cattn_stats<<<dim3(T / 16, H, B), dim3(32), 0, stream>>>(QHB, KHB, SMAX, SINV, T, TR, scale);
        cattn_out<<<dim3(TR / 16, H, B), dim3(32), 0, stream>>>(QHB, KHB, VHB, SMAX, SINV, SCR,
                                                                T, TR, C, scale);
        CVT(SCRB, SCR, (size_t)B * TR * C);
        CVTW(WBF, Pf(base + 19), C, C, 0);
        GEMM(xalt, SCRB, WBF, B * TR, C, C, Pf(base + 20), nullptr, 0);
        { float* tmp = x; x = xalt; xalt = tmp; }
        T = TR; BT = B * T;

        // ---- compress block ff: x += ffwd(ln2(x)) ----
        LNK(LNB, x, Pf(base + 21), Pf(base + 22), BT);
        CVTW(WBF, Pf(base + 23), C, HID, 0);
        GEMM(SCR, LNB, WBF, BT, HID, C, Pf(base + 24), nullptr, 1);
        CVT(SCRB, SCR, (size_t)BT * HID);
        CVTW(WBF, Pf(base + 25), HID, C, 0);
        GEMM(x, SCRB, WBF, BT, C, HID, Pf(base + 26), x, 0);
    }

    // ---- final LN + tied logits (tok is [VOC][C]; need [C][VOC] -> trans=1) ----
    int BT = B * T;   // T == 2048 here
    LNK(LNB, x, Pf(114), Pf(115), BT);
    CVTW(WBF, Pf(0), C, VOC, 1);
    GEMM((float*)d_out, LNB, WBF, BT, VOC, C, Pf(116), nullptr, 0);
}
